// AttentionBlock_19473381720082
// MI455X (gfx1250) — matrix operand order repaired and checked by reading, not compiled, not run on hardware
//
#include <hip/hip_runtime.h>
#include <stdint.h>

// ---------------------------------------------------------------------------
// AttentionBlock for MI455X (gfx1250, wave32, WMMA 16x16x32 f16).
// B=8, C=64, L=2048, H=4, CH=16, G=4.
// All GEMMs (QKV, QK^T, PV, proj) run on v_wmma_f32_16x16x32_f16.
// Attention is flash-style: 16-row Q tiles, streaming 32-column K/V chunks,
// online softmax; the 512MB score tensor is never materialized.
// ---------------------------------------------------------------------------

typedef __attribute__((ext_vector_type(16))) _Float16 v16h;
typedef __attribute__((ext_vector_type(8)))  float    v8f;

#define B_  8
#define C_  64
#define L_  2048
#define H_  4
#define CH_ 16
#define G_  4
#define O3_ (3 * C_)          // 192
#define EPS_ 1e-5f

// 16-bit A-matrix 16x32 as this code fills it (not the ISA table k order): lane = 16*half + m.
// v16h element i (VGPR i/2, halfword i%2) holds K index:
//   K = i + 16*half (contiguous, matching the B operand fill)
__device__ __forceinline__ int a_k_of(int i, int half) {
  return i + 16 * half;
}
// 16-bit B-matrix 32x16: lane = 16*half + n; element i -> K = i + 16*half.
// 32-bit C/D 16x16:      element r -> row = r + 8*half, col = lane & 15.

__device__ __forceinline__ v8f wmma16(v16h a, v16h b, v8f c) {
  // (neg_a, A, neg_b, B, c_mod, C, reuse_a, reuse_b)
  return __builtin_amdgcn_wmma_f32_16x16x32_f16(false, a, false, b, (short)0, c,
                                                false, false);
}

// --- 0: convert weights f32 -> f16 once -----------------------------------
__global__ void k_cvt_w(const float* __restrict__ qw, const float* __restrict__ pw,
                        _Float16* __restrict__ qh, _Float16* __restrict__ ph) {
  int i = blockIdx.x * blockDim.x + threadIdx.x;
  if (i < O3_ * C_) qh[i] = (_Float16)qw[i];
  if (i < C_ * C_)  ph[i] = (_Float16)pw[i];
}

// --- 1: GroupNorm, one block per (b, g); write normalized x as f16 --------
__global__ void k_gn(const float* __restrict__ x, const float* __restrict__ gw,
                     const float* __restrict__ gb, _Float16* __restrict__ xn) {
  const int b = blockIdx.x >> 2, g = blockIdx.x & 3;
  const int GC = C_ / G_;                 // 16 channels per group
  const int N  = GC * L_;                 // 32768 elements per (b,g)
  const size_t base = ((size_t)b * C_ + (size_t)g * GC) * L_;
  const float* xp = x + base;
  float s = 0.f, q = 0.f;
  for (int i = threadIdx.x; i < N; i += 256) { float v = xp[i]; s += v; q += v * v; }
  __shared__ float rs[256], rq[256];
  rs[threadIdx.x] = s; rq[threadIdx.x] = q;
  __syncthreads();
  for (int off = 128; off > 0; off >>= 1) {
    if ((int)threadIdx.x < off) {
      rs[threadIdx.x] += rs[threadIdx.x + off];
      rq[threadIdx.x] += rq[threadIdx.x + off];
    }
    __syncthreads();
  }
  const float mean = rs[0] / (float)N;
  const float var  = rq[0] / (float)N - mean * mean;
  const float inv  = rsqrtf(var + EPS_);
  for (int i = threadIdx.x; i < N; i += 256) {
    const int ch = g * GC + i / L_;
    xn[base + i] = (_Float16)((xp[i] - mean) * inv * gw[ch] + gb[ch]);
  }
}

// --- 2: QKV GEMM, out[b,o,l] = qkv_w[o,:] . xn[b,:,l]; q pre-scaled 0.25 --
__global__ void k_qkv(const _Float16* __restrict__ xn, const _Float16* __restrict__ wq,
                      const float* __restrict__ bias, _Float16* __restrict__ qkv) {
  const int lane = threadIdx.x & 31;
  const int wid  = (int)((blockIdx.x * blockDim.x + threadIdx.x) >> 5);
  const int nt = wid & 127;
  const int t2 = wid >> 7;
  const int mt = t2 % 12;                 // 192/16 output-row tiles
  const int b  = t2 / 12;
  const int half = lane >> 4, lm = lane & 15;
  const int o0 = mt * 16, n0 = nt * 16;
  v8f acc = {};
#pragma unroll
  for (int kb = 0; kb < 2; ++kb) {        // K = 64 = 2 x 32
    const int k0 = kb * 32;
    v16h A, Bx;
#pragma unroll
    for (int i = 0; i < 16; ++i) {
      A[i]  = wq[(o0 + lm) * C_ + k0 + a_k_of(i, half)];
      Bx[i] = xn[((size_t)b * C_ + (size_t)(k0 + i + 16 * half)) * L_ + n0 + lm];
    }
    acc = wmma16(A, Bx, acc);
  }
#pragma unroll
  for (int r = 0; r < 8; ++r) {
    const int o = o0 + r + 8 * half;
    float v = acc[r] + bias[o];
    if (o < C_) v *= 0.25f;               // fold (1/sqrt(sqrt(CH)))^2 into Q
    qkv[((size_t)b * O3_ + o) * L_ + n0 + lm] = (_Float16)v;
  }
}

// --- 3: flash attention; one wave per (head, 16-row Q tile) ---------------
__global__ void k_flash(const _Float16* __restrict__ qkv, _Float16* __restrict__ aout) {
  const int lane = threadIdx.x & 31;
  const int wib  = threadIdx.x >> 5;
  const int wid  = blockIdx.x * 8 + wib;
  const int rt = wid & 127;               // query-row tile (2048/16)
  const int bh = wid >> 7;                // 0..31
  const int b = bh >> 2, h = bh & 3;
  const int half = lane >> 4, lm = lane & 15;
  __shared__ _Float16 smem[8][16 * 40];   // per-wave P tile, padded row stride
  _Float16* P = &smem[wib][0];
  const _Float16* qb = qkv + ((size_t)b * O3_ + (size_t)h * CH_) * L_;
  const _Float16* kb = qb + (size_t)C_ * L_;
  const _Float16* vb = qb + (size_t)(2 * C_) * L_;

  // Q as A-matrix: rows = 16 query positions, K = channel (16, zero-padded to 32)
  v16h Aq;
#pragma unroll
  for (int i = 0; i < 16; ++i) {
    const int k = a_k_of(i, half);
    Aq[i] = (k < CH_) ? qb[(size_t)k * L_ + rt * 16 + lm] : (_Float16)0.f;
  }
  v8f acc = {};
  float mrow[8], lsum[8];
#pragma unroll
  for (int r = 0; r < 8; ++r) { mrow[r] = -1e30f; lsum[r] = 0.f; }

  for (int t = 0; t < L_ / 32; ++t) {     // 64 chunks of 32 key columns
    const int c0 = t * 32;
    // two K B-tiles (channel x 16 columns, zero-padded K rows 16..31)
    v16h B0, B1;
#pragma unroll
    for (int i = 0; i < 16; ++i) {
      const int kk = i + 16 * half;
      B0[i] = (kk < CH_) ? kb[(size_t)kk * L_ + c0 + lm]      : (_Float16)0.f;
      B1[i] = (kk < CH_) ? kb[(size_t)kk * L_ + c0 + 16 + lm] : (_Float16)0.f;
    }
    if (t + 1 < L_ / 32)
      __builtin_prefetch(kb + (size_t)lm * L_ + c0 + 32, 0, 1);  // global_prefetch_b8
    v8f s0 = {}, s1 = {};
    s0 = wmma16(Aq, B0, s0);
    s1 = wmma16(Aq, B1, s1);

    // online softmax over the 32 new columns; rows live as C-layout elements
#pragma unroll
    for (int r = 0; r < 8; ++r) {
      float mx = fmaxf(s0[r], s1[r]);
#pragma unroll
      for (int mk = 1; mk < 16; mk <<= 1) mx = fmaxf(mx, __shfl_xor(mx, mk, 32));
      const float nm = fmaxf(mrow[r], mx);
      const float p0 = __expf(s0[r] - nm);
      const float p1 = __expf(s1[r] - nm);
      const int m = r + 8 * half;
      P[m * 40 + lm]      = (_Float16)p0;  // transpose P through LDS (C->A layout)
      P[m * 40 + 16 + lm] = (_Float16)p1;
      float ps = p0 + p1;
#pragma unroll
      for (int mk = 1; mk < 16; mk <<= 1) ps += __shfl_xor(ps, mk, 32);
      const float corr = __expf(mrow[r] - nm);
      lsum[r] = lsum[r] * corr + ps;
      acc[r] *= corr;
      mrow[r] = nm;
    }
    __builtin_amdgcn_wave_barrier();       // DS in-order per wave; pin scheduling
    v16h Ap, Bv;
#pragma unroll
    for (int i = 0; i < 16; ++i) {
      Ap[i] = P[lm * 40 + a_k_of(i, half)];                       // P as A (16x32)
      Bv[i] = vb[(size_t)lm * L_ + c0 + i + 16 * half];           // V^T as B (32x16)
    }
    __builtin_amdgcn_wave_barrier();
    acc = wmma16(Ap, Bv, acc);             // out[m][c] += P[m][n] * v[c][n]
  }
#pragma unroll
  for (int r = 0; r < 8; ++r) {
    const int l = rt * 16 + r + 8 * half;
    aout[((size_t)b * C_ + (size_t)h * CH_ + lm) * L_ + l] = (_Float16)(acc[r] / lsum[r]);
  }
}

// --- 4: proj GEMM + bias + residual, f32 output ---------------------------
__global__ void k_proj(const _Float16* __restrict__ ah, const _Float16* __restrict__ wp,
                       const float* __restrict__ pb, const float* __restrict__ x,
                       float* __restrict__ out) {
  const int lane = threadIdx.x & 31;
  const int wid  = (int)((blockIdx.x * blockDim.x + threadIdx.x) >> 5);
  const int nt = wid & 127;
  const int t2 = wid >> 7;
  const int mt = t2 & 3;
  const int b  = t2 >> 2;
  const int half = lane >> 4, lm = lane & 15;
  const int o0 = mt * 16, n0 = nt * 16;
  v8f acc = {};
#pragma unroll
  for (int kb2 = 0; kb2 < 2; ++kb2) {
    const int k0 = kb2 * 32;
    v16h A, Ba;
#pragma unroll
    for (int i = 0; i < 16; ++i) {
      A[i]  = wp[(o0 + lm) * C_ + k0 + a_k_of(i, half)];
      Ba[i] = ah[((size_t)b * C_ + (size_t)(k0 + i + 16 * half)) * L_ + n0 + lm];
    }
    acc = wmma16(A, Ba, acc);
  }
#pragma unroll
  for (int r = 0; r < 8; ++r) {
    const int o = o0 + r + 8 * half;
    const size_t idx = ((size_t)b * C_ + o) * L_ + n0 + lm;
    out[idx] = x[idx] + acc[r] + pb[o];
  }
}

// ---------------------------------------------------------------------------
extern "C" void kernel_launch(void* const* d_in, const int* in_sizes, int n_in,
                              void* d_out, int out_size, void* d_ws, size_t ws_size,
                              hipStream_t stream) {
  (void)in_sizes; (void)n_in; (void)out_size; (void)ws_size;
  const float* x    = (const float*)d_in[0];
  const float* gn_w = (const float*)d_in[1];
  const float* gn_b = (const float*)d_in[2];
  const float* qkvw = (const float*)d_in[3];
  const float* qkvb = (const float*)d_in[4];
  const float* pw   = (const float*)d_in[5];
  const float* pb   = (const float*)d_in[6];
  float* out = (float*)d_out;

  // workspace layout (~8.03 MB total):
  char* ws = (char*)d_ws;
  _Float16* xn_h  = (_Float16*)ws;                          // 2 MB  [B,C,L] f16
  _Float16* qkv_h = (_Float16*)(ws + (size_t)(2u << 20));   // 6 MB  [B,192,L] f16
  _Float16* wq_h  = (_Float16*)(ws + (size_t)(8u << 20));   // 24 KB
  _Float16* wp_h  = (_Float16*)(ws + (size_t)(8u << 20) + 24576); // 8 KB
  _Float16* a_h   = xn_h;  // xn is dead after k_qkv; reuse its buffer

  k_cvt_w<<<48, 256, 0, stream>>>(qkvw, pw, wq_h, wp_h);
  k_gn<<<B_ * G_, 256, 0, stream>>>(x, gn_w, gn_b, xn_h);
  k_qkv<<<(B_ * 12 * 128) / 8, 256, 0, stream>>>(xn_h, wq_h, qkvb, qkv_h);   // 1536 blocks
  k_flash<<<(B_ * H_ * 128) / 8, 256, 0, stream>>>(qkv_h, a_h);              // 512 blocks
  k_proj<<<(B_ * 4 * 128) / 8, 256, 0, stream>>>(a_h, wp_h, pb, x, out);     // 512 blocks
}
